// RelationalGraphBuilder_21792664060748
// MI455X (gfx1250) — matrix operand order repaired and checked by reading, not compiled, not run on hardware
//
#include <hip/hip_runtime.h>
#include <hip/hip_bf16.h>
#include <stdint.h>

// Sizes from reference: B=1, T=192, H=1024, NH=16, hd=64, 3H=3072
#define T_DIM  192
#define H_DIM  1024
#define NHEAD  16
#define HD     64
#define H3     3072
#define LN_EPS 1e-5f
// edges = softmax(...,-1) so edges.sum(-1)==1 and the additive attention bias is the
// constant 1/(T+1e-6) for every (q,k); softmax shift-invariance makes build_edges dead code.
#define EWC       (1.0f / (192.0f + 1e-6f))
#define SCALE_INV 0.125f   // 1/sqrt(hd)

#define BN        256      // N columns per GEMM block
#define LDSTRIDE  264      // padded bf16 elems per k-row in LDS (512B data + 16B pad)

typedef __attribute__((ext_vector_type(16))) __bf16 v16bf;
typedef __attribute__((ext_vector_type(8)))  float  v8f;
typedef __attribute__((ext_vector_type(4)))  unsigned int u32x4;
typedef __attribute__((ext_vector_type(8)))  int   i32x8;
typedef __attribute__((ext_vector_type(4)))  int   i32x4;

union FragB { v16bf v; uint32_t u[8]; };
union FragF { v8f   v; float    f[8]; };
union U8i   { i32x8 v; uint32_t w[8]; };
union U4i   { i32x4 v; uint32_t w[4]; };
union U4u   { u32x4 v; uint32_t w[4]; };

#define HAS_TDM __has_builtin(__builtin_amdgcn_tensor_load_to_lds)

static __device__ __forceinline__ uint16_t f2bf(float f) {
  uint32_t x = __float_as_uint(f);
  x += 0x7FFFu + ((x >> 16) & 1u);           // round-to-nearest-even
  return (uint16_t)(x >> 16);
}
static __device__ __forceinline__ uint32_t pk2(float a, float b) {
  return (uint32_t)f2bf(a) | ((uint32_t)f2bf(b) << 16);
}

// ---------------- fp32 -> bf16 cast (row-major copy, for activations) ----------------
__global__ void cast_f32_bf16_k(const float* __restrict__ in,
                                uint16_t* __restrict__ out, int n) {
  int i = blockIdx.x * blockDim.x + threadIdx.x;
  if (i < n) out[i] = f2bf(in[i]);
}

// ------------- fp32 [N,K] -> bf16 [K,N] transpose-cast (weights, once per launch) -------------
// 64x64 tiles via LDS; coalesced loads and stores.
__global__ void castT_f32_bf16_k(const float* __restrict__ in,  // [Nr, Kc]
                                 uint16_t* __restrict__ out,    // [Kc, Nr]
                                 int Nr, int Kc) {
  __shared__ uint16_t tile[64][72];
  const int tn = blockIdx.x * 64;      // n block
  const int tk = blockIdx.y * 64;      // k block
  const int t  = threadIdx.x;          // 256
  const int r  = t >> 2;               // 0..63
  const int cc = (t & 3) * 16;         // 0,16,32,48
  const float* src = in + (size_t)(tn + r) * Kc + tk + cc;
#pragma unroll
  for (int e = 0; e < 16; ++e) tile[r][cc + e] = f2bf(src[e]);
  __syncthreads();
  uint16_t* dst = out + (size_t)(tk + r) * Nr + tn + cc;
#pragma unroll
  for (int e = 0; e < 16; ++e) dst[e] = tile[cc + e][r];
}

// ---------------- GEMM: C[M,N] = A[M,K](bf16) * Wt[K,N](bf16) + bias ----------------
// block = 256 (8 waves); each wave owns a 16x32 strip; block covers 16(M) x 256(N).
// Weight tiles are staged global->LDS with the Tensor Data Mover (double-buffered,
// issued by wave 0, waited with s_wait_tensorcnt) when the builtin exists.
__global__ void gemm_bf16_wmma(const uint16_t* __restrict__ A,   // [M,K]
                               const uint16_t* __restrict__ Wt,  // [K,N] pre-transposed
                               const float* __restrict__ bias,
                               float* __restrict__ C, int N, int K) {
  __shared__ uint16_t sW[2][32 * LDSTRIDE];  // 2 x 16.5 KB, [k][n] padded
  const int tid  = threadIdx.x;
  const int wave = tid >> 5, lane = tid & 31;
  const int hf   = lane >> 4, lrow = lane & 15;
  const int m_base  = blockIdx.y * 16;
  const int n_block = blockIdx.x * BN;
  const uint16_t* arow = A + (size_t)(m_base + lrow) * K;

  const int skk = tid >> 3;            // 0..31 (manual staging row)
  const int snc = (tid & 7) * 32;      // 0..224 (manual staging col chunk)

  auto stage = [&](int buf, int k0) {
#if HAS_TDM
    if (wave == 0) {
      const uint64_t gaddr = (uint64_t)(uintptr_t)(Wt + (size_t)k0 * N + n_block);
      const uint32_t laddr = (uint32_t)(uintptr_t)(&sW[buf][0]);  // low 32 bits = LDS offset
      const uint32_t td0   = (uint32_t)(N >> 1);                  // dims in 4B units
      U4u g0; U8i g1; U4i gz0, gz1;
      g0.w[0] = 1u;                               // count=1, user descriptor
      g0.w[1] = laddr;                            // lds_addr
      g0.w[2] = (uint32_t)gaddr;                  // global_addr[31:0]
      g0.w[3] = (uint32_t)((gaddr >> 32) & 0x1FFFFFFu) | (2u << 30);  // addr[56:32], type=2
      g1.w[0] = (2u << 16)  |                     // data_size = 4B
                (1u << 20)  |                     // pad_enable
                (6u << 22)  |                     // pad_interval: 128 dwords (512B row)
                (3u << 25);                       // pad_amount: 4 dwords (16B -> stride 528B)
      g1.w[1] = td0 << 16;                        // tensor_dim0 low16 at bits[63:48]
      g1.w[2] = (td0 >> 16) | ((uint32_t)K << 16);        // dim0 hi | tensor_dim1 lo
      g1.w[3] = ((uint32_t)K >> 16) | (128u << 16);       // dim1 hi | tile_dim0 = 512B/4
      g1.w[4] = 32u;                              // tile_dim1 = 32 rows (tile_dim2 = 0)
      g1.w[5] = td0;                              // tensor_dim0_stride low32
      g1.w[6] = 0u; g1.w[7] = 0u;
      gz0.w[0] = gz0.w[1] = gz0.w[2] = gz0.w[3] = 0;
      gz1 = gz0;
#if __clang_major__ >= 23
      U8i gz2; for (int q = 0; q < 8; ++q) gz2.w[q] = 0;
      __builtin_amdgcn_tensor_load_to_lds(g0.v, g1.v, gz0.v, gz1.v, gz2.v, 0);
#else
      __builtin_amdgcn_tensor_load_to_lds(g0.v, g1.v, gz0.v, gz1.v, 0);
#endif
    }
#else
    const uint16_t* src = Wt + (size_t)(k0 + skk) * N + n_block + snc;
    uint16_t* dst = &sW[buf][skk * LDSTRIDE + snc];
    uint4 v0 = *(const uint4*)(src);
    uint4 v1 = *(const uint4*)(src + 8);
    uint4 v2 = *(const uint4*)(src + 16);
    uint4 v3 = *(const uint4*)(src + 24);
    *(uint4*)(dst)      = v0;
    *(uint4*)(dst + 8)  = v1;
    *(uint4*)(dst + 16) = v2;
    *(uint4*)(dst + 24) = v3;
#endif
  };
  auto stage_fence = [&]() {
#if HAS_TDM
    if (wave == 0) __builtin_amdgcn_s_wait_tensorcnt(0);
#endif
    __syncthreads();
  };

  FragF acc0, acc1;
#pragma unroll
  for (int i = 0; i < 8; ++i) { acc0.f[i] = 0.0f; acc1.f[i] = 0.0f; }

  stage(0, 0);
  stage_fence();
  int buf = 0;
  for (int k0 = 0; k0 < K; k0 += 32) {
    if (k0 + 32 < K) stage(buf ^ 1, k0 + 32);     // async fill of back buffer

    FragB af, b0, b1;                             // A frag: lanes = M rows
    *(uint4*)&af.u[0] = *(const uint4*)(arow + k0 + hf * 8);
    *(uint4*)&af.u[4] = *(const uint4*)(arow + k0 + 16 + hf * 8);
    const uint16_t* bp = &sW[buf][wave * 32 + lrow];
#pragma unroll
    for (int i = 0; i < 8; ++i) {
      const int kk = (i < 4) ? (hf * 8 + 2 * i) : (16 + hf * 8 + 2 * (i - 4));
      b0.u[i] = (uint32_t)bp[kk * LDSTRIDE] | ((uint32_t)bp[(kk + 1) * LDSTRIDE] << 16);
      b1.u[i] = (uint32_t)bp[kk * LDSTRIDE + 16] | ((uint32_t)bp[(kk + 1) * LDSTRIDE + 16] << 16);
    }
    acc0.v = __builtin_amdgcn_wmma_f32_16x16x32_bf16(false, af.v, false, b0.v,
                                                     (short)0, acc0.v, false, false);
    acc1.v = __builtin_amdgcn_wmma_f32_16x16x32_bf16(false, af.v, false, b1.v,
                                                     (short)0, acc1.v, false, false);
    if (k0 + 32 < K) { stage_fence(); buf ^= 1; }
  }

  const int n0 = n_block + wave * 32 + lrow;
  const int n1 = n0 + 16;
  const float bn0 = bias[n0], bn1 = bias[n1];
#pragma unroll
  for (int i = 0; i < 8; ++i) {
    const int m = m_base + i + 8 * hf;
    C[(size_t)m * N + n0] = acc0.f[i] + bn0;
    C[(size_t)m * N + n1] = acc1.f[i] + bn1;
  }
}

// ---------------- fused attention per head: S=QK^T/8 + c, softmax, O=PV ----------------
__global__ void attn_wmma(const float* __restrict__ qkv,
                          uint16_t* __restrict__ attn_out_bf) {
  __shared__ uint16_t sKt[64 * 200];         // K^T: [d][key], padded stride 200
  __shared__ uint16_t sV[192 * 72];          // V:   [key][d], padded stride 72
  __shared__ uint16_t sP[8 * 16 * 40];       // per-wave P chunk [16][40]
  const int h    = blockIdx.x;
  const int tid  = threadIdx.x;
  const int wave = tid >> 5, lane = tid & 31;
  const int hf   = lane >> 4, lrow = lane & 15;

  for (int idx = tid; idx < 192 * 64; idx += 256) {
    int key = idx >> 6, d = idx & 63;
    const float* base = qkv + (size_t)key * H3 + h * HD + d;
    sKt[d * 200 + key] = f2bf(base[H_DIM]);
    sV[key * 72 + d]   = f2bf(base[2 * H_DIM]);
  }
  __syncthreads();

  uint16_t* Pw = sP + wave * 16 * 40;

  for (int qt = wave; qt < 12; qt += 8) {
    const float* qrow = qkv + (size_t)(qt * 16 + lrow) * H3 + h * HD;
    FragB aq[2];
#pragma unroll
    for (int d2 = 0; d2 < 2; ++d2) {
      int b1 = d2 * 32 + hf * 8;
      float4 f0 = *(const float4*)(qrow + b1);
      float4 f1 = *(const float4*)(qrow + b1 + 4);
      aq[d2].u[0] = pk2(f0.x, f0.y); aq[d2].u[1] = pk2(f0.z, f0.w);
      aq[d2].u[2] = pk2(f1.x, f1.y); aq[d2].u[3] = pk2(f1.z, f1.w);
      int b2 = d2 * 32 + 16 + hf * 8;
      float4 f2 = *(const float4*)(qrow + b2);
      float4 f3 = *(const float4*)(qrow + b2 + 4);
      aq[d2].u[4] = pk2(f2.x, f2.y); aq[d2].u[5] = pk2(f2.z, f2.w);
      aq[d2].u[6] = pk2(f3.x, f3.y); aq[d2].u[7] = pk2(f3.z, f3.w);
    }

    float s[12][8];
#pragma unroll
    for (int kt = 0; kt < 12; ++kt) {
      FragF acc;
#pragma unroll
      for (int i = 0; i < 8; ++i) acc.f[i] = 0.0f;
#pragma unroll
      for (int d2 = 0; d2 < 2; ++d2) {
        FragB bk;
        const uint16_t* kp = sKt + (d2 * 32) * 200 + kt * 16 + lrow;
#pragma unroll
        for (int i = 0; i < 8; ++i) {
          const int kk = (i < 4) ? (hf * 8 + 2 * i) : (16 + hf * 8 + 2 * (i - 4));
          bk.u[i] = (uint32_t)kp[kk * 200] | ((uint32_t)kp[(kk + 1) * 200] << 16);
        }
        acc.v = __builtin_amdgcn_wmma_f32_16x16x32_bf16(false, aq[d2].v, false, bk.v,
                                                        (short)0, acc.v, false, false);
      }
#pragma unroll
      for (int i = 0; i < 8; ++i) s[kt][i] = acc.f[i] * SCALE_INV + EWC;
    }

#pragma unroll
    for (int i = 0; i < 8; ++i) {
      float mx = -3.4e38f;
#pragma unroll
      for (int kt = 0; kt < 12; ++kt) mx = fmaxf(mx, s[kt][i]);
      mx = fmaxf(mx, __shfl_xor(mx, 1, 32));
      mx = fmaxf(mx, __shfl_xor(mx, 2, 32));
      mx = fmaxf(mx, __shfl_xor(mx, 4, 32));
      mx = fmaxf(mx, __shfl_xor(mx, 8, 32));
      float sum = 0.0f;
#pragma unroll
      for (int kt = 0; kt < 12; ++kt) { float e = __expf(s[kt][i] - mx); s[kt][i] = e; sum += e; }
      sum += __shfl_xor(sum, 1, 32);
      sum += __shfl_xor(sum, 2, 32);
      sum += __shfl_xor(sum, 4, 32);
      sum += __shfl_xor(sum, 8, 32);
      float inv = 1.0f / sum;
#pragma unroll
      for (int kt = 0; kt < 12; ++kt) s[kt][i] *= inv;
    }

    FragF oacc[4];
#pragma unroll
    for (int nt = 0; nt < 4; ++nt)
#pragma unroll
      for (int i = 0; i < 8; ++i) oacc[nt].f[i] = 0.0f;

    for (int j = 0; j < 6; ++j) {
#pragma unroll
      for (int t2 = 0; t2 < 2; ++t2) {
        int kt = 2 * j + t2;
#pragma unroll
        for (int i = 0; i < 8; ++i)
          Pw[(i + 8 * hf) * 40 + t2 * 16 + lrow] = f2bf(s[kt][i]);
      }
      __asm__ volatile("s_wait_dscnt 0" ::: "memory");  // wave-local LDS RAW fence
      FragB ap;
      *(uint4*)&ap.u[0] = *(const uint4*)(Pw + lrow * 40 + hf * 8);
      *(uint4*)&ap.u[4] = *(const uint4*)(Pw + lrow * 40 + 16 + hf * 8);
#pragma unroll
      for (int nt = 0; nt < 4; ++nt) {
        FragB bv;
        const uint16_t* vp = sV + (j * 32) * 72 + nt * 16 + lrow;
#pragma unroll
        for (int i = 0; i < 8; ++i) {
          const int kk = (i < 4) ? (hf * 8 + 2 * i) : (16 + hf * 8 + 2 * (i - 4));
          bv.u[i] = (uint32_t)vp[kk * 72] | ((uint32_t)vp[(kk + 1) * 72] << 16);
        }
        oacc[nt].v = __builtin_amdgcn_wmma_f32_16x16x32_bf16(false, ap.v, false, bv.v,
                                                             (short)0, oacc[nt].v, false, false);
      }
    }
#pragma unroll
    for (int nt = 0; nt < 4; ++nt)
#pragma unroll
      for (int i = 0; i < 8; ++i) {
        int m = qt * 16 + i + 8 * hf;
        attn_out_bf[(size_t)m * H_DIM + h * HD + nt * 16 + lrow] = f2bf(oacc[nt].f[i]);
      }
  }
}

// ---------------- LayerNorm(proj) + residual; optional bf16 mirror ----------------
__global__ void ln_residual(const float* __restrict__ proj,
                            const float* __restrict__ xprev,
                            const float* __restrict__ g,
                            const float* __restrict__ b,
                            float* __restrict__ xout,
                            uint16_t* __restrict__ xb16) {
  __shared__ float r1[8], r2[8];
  const int row = blockIdx.x, tid = threadIdx.x;
  const int wave = tid >> 5, lane = tid & 31;
  const float* p = proj + (size_t)row * H_DIM;
  float s1 = 0.0f, s2 = 0.0f;
  for (int c = tid; c < H_DIM; c += 256) { float v = p[c]; s1 += v; s2 += v * v; }
#pragma unroll
  for (int off = 16; off >= 1; off >>= 1) {
    s1 += __shfl_xor(s1, off, 32);
    s2 += __shfl_xor(s2, off, 32);
  }
  if (lane == 0) { r1[wave] = s1; r2[wave] = s2; }
  __syncthreads();
  s1 = 0.0f; s2 = 0.0f;
#pragma unroll
  for (int w = 0; w < 8; ++w) { s1 += r1[w]; s2 += r2[w]; }
  const float mean = s1 * (1.0f / H_DIM);
  const float var  = s2 * (1.0f / H_DIM) - mean * mean;
  const float rstd = rsqrtf(var + LN_EPS);
  for (int c = tid; c < H_DIM; c += 256) {
    size_t idx = (size_t)row * H_DIM + c;
    float o = xprev[idx] + (p[c] - mean) * rstd * g[c] + b[c];
    xout[idx] = o;
    if (xb16) xb16[idx] = f2bf(o);
  }
}

extern "C" void kernel_launch(void* const* d_in, const int* in_sizes, int n_in,
                              void* d_out, int out_size, void* d_ws, size_t ws_size,
                              hipStream_t stream) {
  (void)in_sizes; (void)n_in; (void)out_size; (void)ws_size;
  const float* nodes = (const float*)d_in[0];
  const float* in_w  = (const float*)d_in[14];   // [2,3072,1024]
  const float* in_b  = (const float*)d_in[15];   // [2,3072]
  const float* out_w = (const float*)d_in[16];   // [2,1024,1024]
  const float* out_b = (const float*)d_in[17];   // [2,1024]
  const float* ln_g  = (const float*)d_in[18];   // [2,1024]
  const float* ln_b  = (const float*)d_in[19];   // [2,1024]

  char* ws = (char*)d_ws;
  size_t off = 0;
  auto carve = [&](size_t bytes) -> void* {
    void* p = ws + off;
    off = (off + bytes + 255) & ~(size_t)255;
    return p;
  };
  uint16_t* wbin  = (uint16_t*)carve((size_t)2 * H3 * H_DIM * 2);      // bf16 W_in^T [K,N] per layer
  uint16_t* wbout = (uint16_t*)carve((size_t)2 * H_DIM * H_DIM * 2);   // bf16 W_out^T [K,N] per layer
  uint16_t* xb    = (uint16_t*)carve((size_t)T_DIM * H_DIM * 2);       // bf16 activations
  float*    qkv   = (float*)carve((size_t)T_DIM * H3 * 4);
  uint16_t* abf   = (uint16_t*)carve((size_t)T_DIM * H_DIM * 2);       // bf16 attention output
  float*    proj  = (float*)carve((size_t)T_DIM * H_DIM * 4);
  float*    x1    = (float*)carve((size_t)T_DIM * H_DIM * 4);

  // weight repack: fp32 [N,K] -> bf16 [K,N], per layer (transpose makes GEMM tiles TDM-friendly)
  for (int l = 0; l < 2; ++l) {
    castT_f32_bf16_k<<<dim3(H3 / 64, H_DIM / 64), 256, 0, stream>>>(
        in_w + (size_t)l * H3 * H_DIM, wbin + (size_t)l * H3 * H_DIM, H3, H_DIM);
    castT_f32_bf16_k<<<dim3(H_DIM / 64, H_DIM / 64), 256, 0, stream>>>(
        out_w + (size_t)l * H_DIM * H_DIM, wbout + (size_t)l * H_DIM * H_DIM, H_DIM, H_DIM);
  }
  int n3 = T_DIM * H_DIM;
  cast_f32_bf16_k<<<(n3 + 255) / 256, 256, 0, stream>>>(nodes, xb, n3);

  for (int l = 0; l < 2; ++l) {
    gemm_bf16_wmma<<<dim3(H3 / BN, T_DIM / 16), 256, 0, stream>>>(
        xb, wbin + (size_t)l * H3 * H_DIM, in_b + l * H3, qkv, H3, H_DIM);
    attn_wmma<<<NHEAD, 256, 0, stream>>>(qkv, abf);
    gemm_bf16_wmma<<<dim3(H_DIM / BN, T_DIM / 16), 256, 0, stream>>>(
        abf, wbout + (size_t)l * H_DIM * H_DIM, out_b + l * H_DIM, proj, H_DIM, H_DIM);
    ln_residual<<<T_DIM, 256, 0, stream>>>(
        proj, (l == 0) ? nodes : x1, ln_g + l * H_DIM, ln_b + l * H_DIM,
        (l == 0) ? x1 : (float*)d_out, (l == 0) ? xb : (uint16_t*)nullptr);
  }
}
